// CPE_47364899340506
// MI455X (gfx1250) — matrix operand order repaired and checked by reading, not compiled, not run on hardware
//
#include <hip/hip_runtime.h>

typedef _Float16 v8h  __attribute__((ext_vector_type(8)));
typedef _Float16 v16h __attribute__((ext_vector_type(16)));
typedef float    v8f  __attribute__((ext_vector_type(8)));
typedef unsigned int u32x4 __attribute__((ext_vector_type(4)));
typedef int      i32x8 __attribute__((ext_vector_type(8)));
typedef int      i32x4 __attribute__((ext_vector_type(4)));

#if defined(__has_builtin)
#  if __has_builtin(__builtin_amdgcn_tensor_load_to_lds) && __has_builtin(__builtin_amdgcn_s_wait_tensorcnt)
#    define HAVE_TDM 1
#  else
#    define HAVE_TDM 0
#  endif
#else
#  define HAVE_TDM 0
#endif

#define GRID 65           // voxel grid side (BND+1)
#define CDIM 256
#define LDA  264          // A tile LDS row stride in halves (512B data + 16B pad)
#define LDB  40           // B chunk LDS row stride in halves (64B data + 16B pad)

// ---------------- prep kernels ----------------

__global__ void coords_kernel(const float* __restrict__ depth, int* __restrict__ coordp) {
  __shared__ float smin[256], smax[256];
  const int b = blockIdx.x, t = threadIdx.x;
  const float* d = depth + b * 4096;
  float mn = 1e30f, mx = -1e30f;
  for (int i = t; i < 4096; i += 256) { float v = d[i]; mn = fminf(mn, v); mx = fmaxf(mx, v); }
  smin[t] = mn; smax[t] = mx;
  __syncthreads();
  for (int s = 128; s > 0; s >>= 1) {
    if (t < s) { smin[t] = fminf(smin[t], smin[t + s]); smax[t] = fmaxf(smax[t], smax[t + s]); }
    __syncthreads();
  }
  const float zmin = smin[0];
  const float den  = smax[0] - zmin + 1e-8f;
  for (int i = t; i < 4096; i += 256) {
    int row = i >> 6, col = i & 63;
    float xf = (float)col / 63.0f;
    float yf = (float)row / 63.0f;
    float zf = (d[i] - zmin) / den;
    int cx = (int)fminf(fmaxf(rintf(xf * 64.0f), 0.0f), 64.0f);
    int cy = (int)fminf(fmaxf(rintf(yf * 64.0f), 0.0f), 64.0f);
    int cz = (int)fminf(fmaxf(rintf(zf * 64.0f), 0.0f), 64.0f);
    coordp[b * 4097 + 1 + i] = cx | (cy << 8) | (cz << 16);
  }
  if (t == 0) coordp[b * 4097] = 0;  // cls token at (0,0,0)
}

__global__ void init_map_kernel(int* __restrict__ idx_map, int total) {
  int i = blockIdx.x * 256 + threadIdx.x;
  if (i < total) idx_map[i] = -1;
}

__global__ void scatter_kernel(const int* __restrict__ coordp, int* __restrict__ idx_map, int N) {
  int n = blockIdx.x * 256 + threadIdx.x;
  if (n >= N) return;
  int p = coordp[n];
  int x = p & 255, y = (p >> 8) & 255, z = (p >> 16) & 255;
  atomicMax(&idx_map[(x * GRID + y) * GRID + z], n);
}

__global__ void nid_kernel(const int* __restrict__ coordp, const int* __restrict__ idx_map,
                           int* __restrict__ nid_tab, int N) {
  int n = blockIdx.x * 256 + threadIdx.x;
  if (n >= N) return;
  int p = coordp[n];
  int x = p & 255, y = (p >> 8) & 255, z = (p >> 16) & 255;
  #pragma unroll
  for (int k = 0; k < 27; ++k) {
    int dx = k / 9 - 1, dy = (k / 3) % 3 - 1, dz = k % 3 - 1;   // itertools.product order
    int nx = x + dx, ny = y + dy, nz = z + dz;
    int v = -1;
    if ((unsigned)nx <= 64u && (unsigned)ny <= 64u && (unsigned)nz <= 64u)
      v = idx_map[(nx * GRID + ny) * GRID + nz];
    nid_tab[k * N + n] = v;
  }
}

__global__ void cvt_feat_kernel(const float* __restrict__ f, _Float16* __restrict__ fh, int count8) {
  int id = blockIdx.x * 256 + threadIdx.x;
  if (id >= count8) return;
  const float4* s = (const float4*)f + (size_t)id * 2;
  float4 a = s[0], b = s[1];
  v8h o;
  o[0] = (_Float16)a.x; o[1] = (_Float16)a.y; o[2] = (_Float16)a.z; o[3] = (_Float16)a.w;
  o[4] = (_Float16)b.x; o[5] = (_Float16)b.y; o[6] = (_Float16)b.z; o[7] = (_Float16)b.w;
  ((v8h*)fh)[id] = o;
}

// wt[k][n][c] = (f16) weight[k][c][n]  (transpose so B fragments are row-contiguous)
__global__ void cvt_wt_kernel(const float* __restrict__ w, _Float16* __restrict__ wt, int total) {
  int id = blockIdx.x * 256 + threadIdx.x;
  if (id >= total) return;
  int k = id >> 16, rem = id & 65535, n = rem >> 8, c = rem & 255;
  wt[id] = (_Float16)w[(k << 16) + (c << 8) + n];
}

// ---------------- main gathered-GEMM kernel ----------------

__global__ __launch_bounds__(256) void subm_gemm_kernel(
    const _Float16* __restrict__ fh, const _Float16* __restrict__ wt,
    const int* __restrict__ nid_tab, float* __restrict__ out, int N)
{
  __shared__ __align__(16) _Float16 Ash[64 * LDA];    // gathered A tile, 64 x 256 f16 (padded rows)
  __shared__ __align__(16) _Float16 Bsh[256 * LDB];   // W^T K-chunk, 256 x 32 f16 (padded rows)

  const int tid   = threadIdx.x;
  const int r0    = blockIdx.x * 64;
  const int lane  = tid & 31;
  const int wv    = tid >> 5;
  const int strip = wv >> 1;          // 4 strips of 16 rows
  const int half  = wv & 1;           // 2 halves of 128 output cols
  const int m     = lane & 15;
  const int hs    = lane >> 4;

  // gather assignment: 4 threads per A row, 8 interleaved 16B chunks each
  const int grow  = r0 + (tid >> 2);  // row this thread gathers
  const int garow = (tid >> 2) * LDA; // LDS row base (halves)
  const int gch0  = tid & 3;          // chunk phase

  v8f acc[8];
  #pragma unroll
  for (int j = 0; j < 8; ++j) acc[j] = (v8f)0.0f;

  const v8h* fhv = (const v8h*)fh;

  for (int k = 0; k < 27; ++k) {
    // ---- Phase A: gather 64 rows for this tap; one index load per thread ----
    const int* nt = nid_tab + (size_t)k * N;
    int nid = (grow < N) ? nt[grow] : -1;
    if (nid >= 0) {
      const v8h* src = fhv + (size_t)nid * 32;
      #pragma unroll
      for (int i = 0; i < 8; ++i) {
        int ch = gch0 + i * 4;
        *(v8h*)&Ash[garow + ch * 8] = src[ch];
      }
    } else {
      v8h z = (v8h)(_Float16)0.0f;
      #pragma unroll
      for (int i = 0; i < 8; ++i) {
        int ch = gch0 + i * 4;
        *(v8h*)&Ash[garow + ch * 8] = z;
      }
    }
    __syncthreads();

    for (int kk = 0; kk < 8; ++kk) {
      // ---- stage W^T[k][:, kk*32 .. kk*32+31] into LDS via TDM ----
#if HAVE_TDM
      if (tid < 32) {
        unsigned lds = (unsigned)(unsigned long long)(&Bsh[0]);
        unsigned long long ga = (unsigned long long)wt + 2ull * ((size_t)k * 65536 + (size_t)kk * 32);
        u32x4 g0;
        g0[0] = 1u;                                  // count=1 valid descriptor
        g0[1] = lds;                                 // lds_addr
        g0[2] = (unsigned)ga;                        // global_addr[31:0]
        g0[3] = (unsigned)(ga >> 32) | (2u << 30);   // global_addr[56:32] | type=2
        i32x8 g1;
        g1[0] = (int)((3u << 16) | (1u << 20) | (3u << 22) | (3u << 25)); // 8B elems, pad: +16B per 64B
        g1[1] = (int)(8u << 16);      // tensor_dim0 = 8 units (64B row)
        g1[2] = (int)(256u << 16);    // tensor_dim1 = 256 rows
        g1[3] = (int)(8u << 16);      // tile_dim0 = 8 units
        g1[4] = 256;                  // tile_dim1 = 256, tile_dim2 = 0
        g1[5] = 64;                   // tensor_dim0_stride = 64 units (512B)
        g1[6] = 0;
        g1[7] = 0;
        i32x4 gz4; gz4[0] = 0; gz4[1] = 0; gz4[2] = 0; gz4[3] = 0;
        i32x8 gz8;
        #pragma unroll
        for (int i = 0; i < 8; ++i) gz8[i] = 0;
        __builtin_amdgcn_tensor_load_to_lds(g0, g1, gz4, gz4, gz8, 0);
        __builtin_amdgcn_s_wait_tensorcnt(0);
      }
#else
      for (int idx = tid; idx < 2048; idx += 256) {
        int n = idx >> 3, u = idx & 7;
        const uint2* src = (const uint2*)(wt + ((size_t)k << 16) + (n << 8) + kk * 32);
        *(uint2*)&Bsh[n * LDB + u * 4] = src[u];
      }
#endif
      __syncthreads();

      // ---- A fragment (ISA 16-bit A 16x32 layout) ----
      const v8h* ap = (const v8h*)&Ash[(strip * 16 + m) * LDA + kk * 32 + hs * 8];
      v8h alo = ap[0];   // K = hs*8 + 0..7
      v8h ahi = ap[2];   // K = 16 + hs*8 + 0..7
      v16h af;
      #pragma unroll
      for (int i = 0; i < 8; ++i) { af[i] = alo[i]; af[i + 8] = ahi[i]; }

      // ---- 8 B fragments + 8 WMMAs ----
      #pragma unroll
      for (int j = 0; j < 8; ++j) {
        int n = half * 128 + j * 16 + m;              // output column = B column = W^T row
        const v8h* bp = (const v8h*)&Bsh[n * LDB + hs * 8];  // K = hs*8 + 0..7, then 16 + hs*8 + 0..7
        v8h blo = bp[0], bhi = bp[2];
        v16h bf;
        #pragma unroll
        for (int i = 0; i < 8; ++i) { bf[i] = blo[i]; bf[i + 8] = bhi[i]; }
        acc[j] = __builtin_amdgcn_wmma_f32_16x16x32_f16(
            false, af, false, bf, (short)0, acc[j], false, false);
      }
      __syncthreads();
    }
  }

  // ---- store: C/D layout -> VGPR r, lane: M = hs*8 + r, N = m ----
  #pragma unroll
  for (int j = 0; j < 8; ++j) {
    int col = half * 128 + j * 16 + m;
    #pragma unroll
    for (int r = 0; r < 8; ++r) {
      int row = r0 + strip * 16 + hs * 8 + r;
      if (row < N) out[(size_t)row * 256 + col] = acc[j][r];
    }
  }
}

// ---------------- launcher ----------------

static inline size_t rnd256(size_t x) { return (x + 255) & ~(size_t)255; }

extern "C" void kernel_launch(void* const* d_in, const int* in_sizes, int n_in,
                              void* d_out, int out_size, void* d_ws, size_t ws_size,
                              hipStream_t stream) {
  const float* features = (const float*)d_in[0];
  const float* depth    = (const float*)d_in[1];
  const float* weight   = (const float*)d_in[2];
  float* out = (float*)d_out;

  const int N = in_sizes[0] / CDIM;        // 65552
  const int MAP = GRID * GRID * GRID;      // 274625

  char* p = (char*)d_ws;
  int* coordp  = (int*)p;        p += rnd256((size_t)N * 4);
  int* idx_map = (int*)p;        p += rnd256((size_t)MAP * 4);
  int* nid_tab = (int*)p;        p += rnd256((size_t)27 * N * 4);
  _Float16* fh = (_Float16*)p;   p += rnd256((size_t)N * CDIM * 2);
  _Float16* wt = (_Float16*)p;   p += rnd256((size_t)27 * CDIM * CDIM * 2);

  coords_kernel<<<16, 256, 0, stream>>>(depth, coordp);
  init_map_kernel<<<(MAP + 255) / 256, 256, 0, stream>>>(idx_map, MAP);
  scatter_kernel<<<(N + 255) / 256, 256, 0, stream>>>(coordp, idx_map, N);
  nid_kernel<<<(N + 255) / 256, 256, 0, stream>>>(coordp, idx_map, nid_tab, N);

  const int count8 = N * (CDIM / 8);
  cvt_feat_kernel<<<(count8 + 255) / 256, 256, 0, stream>>>(features, fh, count8);
  const int wtot = 27 * CDIM * CDIM;
  cvt_wt_kernel<<<(wtot + 255) / 256, 256, 0, stream>>>(weight, wt, wtot);

  subm_gemm_kernel<<<(N + 63) / 64, 256, 0, stream>>>(fh, wt, nid_tab, out, N);
}
